// DCTAttention_87101936763361
// MI455X (gfx1250) — matrix operand order repaired and checked by reading, not compiled, not run on hardware
//
#include <hip/hip_runtime.h>

// DCT-II -> IDCT over T is an exact mathematical identity => skipped.
// Remaining work: 2048 independent 128x128 self-attention problems,
// done with v_wmma_f32_16x16x32_f16 (f32 accumulate) out of LDS.

typedef _Float16 v16h __attribute__((ext_vector_type(16)));
typedef _Float16 v8h  __attribute__((ext_vector_type(8)));
typedef float    v8f  __attribute__((ext_vector_type(8)));

#define CW      128   // C == W == head rows == head dim
#define STRIDE  136   // padded LDS row stride in halfs (272B: 16B-aligned, bank-spread)
#define NWAVE   8

__device__ __forceinline__ v16h cat8(v8h lo, v8h hi) {
  return __builtin_shufflevector(lo, hi, 0,1,2,3,4,5,6,7,8,9,10,11,12,13,14,15);
}

// A-matrix fragment (16x32 f16): row = given, lane<16 holds K {0..7,16..23},
// lane>=16 holds K {8..15,24..31} of the 32-wide k-chunk (CDNA5 ISA 7.12.2).
__device__ __forceinline__ v16h load_fragA(const _Float16* base, int row, int kbase, int hh) {
  const _Float16* p = base + row * STRIDE + kbase + 8 * hh;
  v8h lo = *(const v8h*)(p);
  v8h hi = *(const v8h*)(p + 16);
  return cat8(lo, hi);
}

// B-matrix fragment (32x16 f16): column = lane%16 -> row `row` of the source,
// lanes 0-15 hold K {0..7,16..23}, lanes 16-31 hold K {8..15,24..31} of the chunk.
__device__ __forceinline__ v16h load_fragB(const _Float16* base, int row, int kbase, int hh) {
  const _Float16* p = base + row * STRIDE + kbase + 8 * hh;
  v8h lo = *(const v8h*)(p);
  v8h hi = *(const v8h*)(p + 16);
  return cat8(lo, hi);
}

__global__ __launch_bounds__(256)
void dct_attn_kernel(const float* __restrict__ X, float* __restrict__ Out) {
  extern __shared__ _Float16 smem[];
  _Float16* sX  = smem;               // [128][STRIDE] row-major f16 X
  _Float16* sXT = sX  + CW * STRIDE;  // [128][STRIDE] transposed f16 X
  _Float16* sP  = sXT + CW * STRIDE;  // [8 waves][16][STRIDE] softmax rows

  const int bt   = blockIdx.x;
  const int tid  = threadIdx.x;
  const int lane = tid & 31;
  const int wave = tid >> 5;
  const int nl   = lane & 15;   // column / row-in-tile index
  const int hh   = lane >> 4;   // half-wave id

  const float* __restrict__ Xbt = X   + (size_t)bt * CW * CW;
  float*       __restrict__ Obt = Out + (size_t)bt * CW * CW;

  // ---- Stage: one HBM read of the tile; convert f32 -> f16, row + transposed ----
  const float4* X4 = (const float4*)Xbt;
  #pragma unroll
  for (int i = 0; i < 16; ++i) {
    int idx = tid + i * 256;       // float4 index; linear elem = idx*4
    int c = idx >> 5;              // (idx*4) >> 7
    int w = (idx & 31) * 4;        // (idx*4) & 127
    float4 v = X4[idx];
    _Float16 h0 = (_Float16)v.x, h1 = (_Float16)v.y;
    _Float16 h2 = (_Float16)v.z, h3 = (_Float16)v.w;
    _Float16* r = sX + c * STRIDE + w;
    r[0] = h0; r[1] = h1; r[2] = h2; r[3] = h3;     // contiguous 8B store
    sXT[(w + 0) * STRIDE + c] = h0;
    sXT[(w + 1) * STRIDE + c] = h1;
    sXT[(w + 2) * STRIDE + c] = h2;
    sXT[(w + 3) * STRIDE + c] = h3;
  }
  __syncthreads();

  // ---- GEMM1: S = X * X^T / sqrt(32); wave owns rows [16*wave, 16*wave+16) ----
  const int m0 = wave * 16;
  v16h a1[4];
  #pragma unroll
  for (int ks = 0; ks < 4; ++ks) a1[ks] = load_fragA(sX, m0 + nl, ks * 32, hh);

  v8f acc[8];
  #pragma unroll
  for (int j = 0; j < 8; ++j) {
    v8f c = {};
    #pragma unroll
    for (int ks = 0; ks < 4; ++ks) {
      // B column n = row (16j+n) of X (since B = X^T)
      v16h b = load_fragB(sX, j * 16 + nl, ks * 32, hh);
      c = __builtin_amdgcn_wmma_f32_16x16x32_f16(false, a1[ks], false, b,
                                                 (short)0, c, false, false);
    }
    acc[j] = c;
  }

  // ---- Softmax over 128 columns, on the C/D layout (row r+8*hh spread over 16 lanes) ----
  const float scale = 0.17677669529663687f;  // 1/sqrt(32)
  _Float16* myP = sP + wave * 16 * STRIDE;
  #pragma unroll
  for (int r = 0; r < 8; ++r) {
    float ls[8];
    float mx = -3.4e38f;
    #pragma unroll
    for (int j = 0; j < 8; ++j) { ls[j] = acc[j][r] * scale; mx = fmaxf(mx, ls[j]); }
    #pragma unroll
    for (int off = 8; off > 0; off >>= 1) mx = fmaxf(mx, __shfl_xor(mx, off, 16));
    float p[8], sum = 0.f;
    #pragma unroll
    for (int j = 0; j < 8; ++j) { p[j] = __expf(ls[j] - mx); sum += p[j]; }
    #pragma unroll
    for (int off = 8; off > 0; off >>= 1) sum += __shfl_xor(sum, off, 16);
    float inv = 1.0f / sum;
    int row = r + 8 * hh;
    #pragma unroll
    for (int j = 0; j < 8; ++j)
      myP[row * STRIDE + j * 16 + nl] = (_Float16)(p[j] * inv);
  }
  // wave-private strip: DS ops are same-wave in-order; make the RAW explicit
  asm volatile("s_wait_dscnt 0x0" ::: "memory");

  // ---- GEMM2: O = P * X (contraction over C), B columns from transposed X ----
  v16h a2[4];
  #pragma unroll
  for (int ks = 0; ks < 4; ++ks) a2[ks] = load_fragA(myP, nl, ks * 32, hh);

  #pragma unroll
  for (int j = 0; j < 8; ++j) {
    v8f o = {};
    #pragma unroll
    for (int ks = 0; ks < 4; ++ks) {
      // B column n = X[:, 16j+n] = row (16j+n) of X^T
      v16h b = load_fragB(sXT, j * 16 + nl, ks * 32, hh);
      o = __builtin_amdgcn_wmma_f32_16x16x32_f16(false, a2[ks], false, b,
                                                 (short)0, o, false, false);
    }
    #pragma unroll
    for (int r = 0; r < 8; ++r)
      Obt[(m0 + r + 8 * hh) * CW + j * 16 + nl] = o[r];
  }
}

extern "C" void kernel_launch(void* const* d_in, const int* in_sizes, int n_in,
                              void* d_out, int out_size, void* d_ws, size_t ws_size,
                              hipStream_t stream) {
  (void)n_in; (void)out_size; (void)d_ws; (void)ws_size;
  const float* X = (const float*)d_in[0];
  float* Out = (float*)d_out;
  int bt = in_sizes[0] / (CW * CW);  // 8*256 = 2048 independent problems
  size_t shmem = (size_t)(2 * CW * STRIDE + NWAVE * 16 * STRIDE) * sizeof(_Float16); // ~102 KB
  dct_attn_kernel<<<bt, 256, shmem, stream>>>(X, Out);
}
